// LocalClosedFormResidualHead_6511170421163
// MI455X (gfx1250) — matrix operand order repaired and checked by reading, not compiled, not run on hardware
//
#include <hip/hip_runtime.h>
#include <math.h>

// ---------------------------------------------------------------------------
// LocalClosedFormResidualHead on MI455X (gfx1250, wave32, WMMA)
//
// B=8192, C=512, P=8, D=512. Dominant cost: g[c,p,b] = proto . xn  (M=4096,
// N=8192, K=512 GEMM, 34.4 GFLOP) done with v_wmma_f32_16x16x32_bf16.
// prior GEMM eliminated (centers = mean_p proto => x.centers = mean_p h).
// quad folded: a^T G a = a^T g - lam*|a|^2  since A a = g, G = A - lam I.
// Round 3: global->LDS staging now uses gfx1250 async copies
// (global_load_async_to_lds_b128 / ASYNCcnt / s_wait_asynccnt) instead of
// VGPR round-trips; double-buffered LDS, one barrier per K-chunk.
// ---------------------------------------------------------------------------

typedef __attribute__((ext_vector_type(16))) __bf16          v16bf;
typedef __attribute__((ext_vector_type(8)))  float           v8f;
typedef __attribute__((ext_vector_type(8)))  unsigned short  v8u16;

#define BN 8192
#define CN 512
#define PN 8
#define DN 512
#define RIDGE   0.1f
#define ROUT_T  0.07f
#define PRIOR_T 0.07f
#define NORM_EPS 1e-8f

#define LDA 72                    // padded bf16 row stride in LDS
#define BUFSZ ((64 + 256) * LDA)  // one ping-pong buffer, in bf16 elems
#define SMEM_BYTES (2 * BUFSZ * 2)

static __device__ __forceinline__ unsigned short f2bf(float f) {
    union { float f; unsigned u; } x; x.f = f;
    unsigned r = (x.u + 0x7FFFu + ((x.u >> 16) & 1u)) >> 16;   // RNE
    return (unsigned short)r;
}

static __device__ __forceinline__ v16bf mkfrag(const unsigned short* lo,
                                               const unsigned short* hi) {
    union { v8u16 u[2]; v16bf b; } cv;
    cv.u[0] = *(const v8u16*)lo;
    cv.u[1] = *(const v8u16*)hi;
    return cv.b;
}

// 16B global -> LDS async copy. lds = byte offset within workgroup LDS
// (low 32 bits of the generic pointer: flat addr[31:0] is the LDS offset,
// hardware adds LDS_BASE). Tracked by ASYNCcnt.
static __device__ __forceinline__ void async_cp16(unsigned lds,
                                                  const unsigned short* g) {
    asm volatile("global_load_async_to_lds_b128 %0, %1, off"
                 :: "v"(lds), "v"(g)
                 : "memory");
}

static __device__ __forceinline__ void wait_async0() {
    asm volatile("s_wait_asynccnt 0x0" ::: "memory");
}

// ---------------------------------------------------------------------------
// Kernel 1: row L2-normalize x -> bf16, and store |xn|^2. One wave per row.
// ---------------------------------------------------------------------------
__global__ __launch_bounds__(256) void k_norm(const float* __restrict__ x,
                                              unsigned short* __restrict__ xnB,
                                              float* __restrict__ xsq) {
    int row  = blockIdx.x * 8 + (threadIdx.x >> 5);
    int lane = threadIdx.x & 31;
    const float* xr = x + (size_t)row * DN;
    float v[16];
    float s = 0.f;
#pragma unroll
    for (int i = 0; i < 16; i++) { v[i] = xr[lane + 32 * i]; s += v[i] * v[i]; }
#pragma unroll
    for (int m = 16; m >= 1; m >>= 1) s += __shfl_xor(s, m, 32);
    float scale = 1.f / (sqrtf(s) + NORM_EPS);
    unsigned short* orow = xnB + (size_t)row * DN;
#pragma unroll
    for (int i = 0; i < 16; i++) orow[lane + 32 * i] = f2bf(v[i] * scale);
    if (lane == 0) xsq[row] = s * scale * scale;
}

// ---------------------------------------------------------------------------
// Kernel 2: fp32 -> bf16 convert (prototypes), 4 elems/thread.
// ---------------------------------------------------------------------------
__global__ __launch_bounds__(256) void k_cvt(const float* __restrict__ p,
                                             unsigned short* __restrict__ o) {
    int i = blockIdx.x * blockDim.x + threadIdx.x;     // over n/4 float4s
    float4 v = ((const float4*)p)[i];
    int idx = i * 4;
    o[idx + 0] = f2bf(v.x); o[idx + 1] = f2bf(v.y);
    o[idx + 2] = f2bf(v.z); o[idx + 3] = f2bf(v.w);
}

// ---------------------------------------------------------------------------
// Kernel 3: per-class A^{-1} = (S S^T + lam I)^{-1}, fp32. 64 thr/class.
// ---------------------------------------------------------------------------
__global__ __launch_bounds__(64) void k_ainv(const float* __restrict__ proto,
                                             float* __restrict__ Ainv) {
    int c = blockIdx.x;
    __shared__ float G[PN][PN];
    int t = threadIdx.x;
    int p = t >> 3, q = t & 7;
    const float* rp = proto + ((size_t)c * PN + p) * DN;
    const float* rq = proto + ((size_t)c * PN + q) * DN;
    float s = 0.f;
    for (int d = 0; d < DN; d++) s += rp[d] * rq[d];
    G[p][q] = s + (p == q ? RIDGE : 0.f);
    __syncthreads();
    if (t == 0) {
        float M[PN][2 * PN];
#pragma unroll
        for (int i = 0; i < PN; i++)
#pragma unroll
            for (int j = 0; j < PN; j++) {
                M[i][j] = G[i][j];
                M[i][PN + j] = (i == j) ? 1.f : 0.f;
            }
        // Gauss-Jordan; SPD + ridge => pivots are safe without row swaps.
#pragma unroll
        for (int k = 0; k < PN; k++) {
            float inv = 1.f / M[k][k];
#pragma unroll
            for (int j = 0; j < 2 * PN; j++) M[k][j] *= inv;
#pragma unroll
            for (int i = 0; i < PN; i++) {
                if (i == k) continue;
                float f = M[i][k];
#pragma unroll
                for (int j = 0; j < 2 * PN; j++) M[i][j] -= f * M[k][j];
            }
        }
        float* o = Ainv + (size_t)c * 64;
#pragma unroll
        for (int i = 0; i < PN; i++)
#pragma unroll
            for (int j = 0; j < PN; j++) o[i * PN + j] = M[i][PN + j];
    }
}

// ---------------------------------------------------------------------------
// Kernel 4: bf16 WMMA GEMM + fused dual-ridge epilogue.
// Block = 256 threads (8 waves), macro-tile 64(M) x 256(N), K chunk 64,
// double-buffered LDS filled by async global->LDS copies.
// Wave (wm,wn) owns a 32x64 region = 2x4 wmma tiles.
// C/D VGPR layout => lane (l%16 = b, l/16 = which class of the tile) holds
// the full h[p]=acc[p] vector, so the solve epilogue is purely per-lane.
// ---------------------------------------------------------------------------
static __device__ __forceinline__ void
issue_chunk(const unsigned short* __restrict__ protoB,
            const unsigned short* __restrict__ xnB,
            int Mb, int Nb, int r, int cq, int k0,
            unsigned ldsA, unsigned ldsB) {
    const unsigned short* ga = protoB + (size_t)(Mb + r) * DN + k0 + cq;
    async_cp16(ldsA + 2u * (r * LDA + cq), ga);
    async_cp16(ldsA + 2u * (r * LDA + cq + 8), ga + 8);
#pragma unroll
    for (int i = 0; i < 4; i++) {
        int rr = r + 64 * i;
        const unsigned short* gb =
            xnB + (size_t)(Nb + rr) * DN + k0 + cq;
        async_cp16(ldsB + 2u * (rr * LDA + cq), gb);
        async_cp16(ldsB + 2u * (rr * LDA + cq + 8), gb + 8);
    }
}

__global__ __launch_bounds__(256) void k_gemm(
    const unsigned short* __restrict__ protoB,  // [4096][512] bf16
    const unsigned short* __restrict__ xnB,     // [8192][512] bf16
    const float* __restrict__ xsq,              // [8192]
    const float* __restrict__ Ainv,             // [512][64]
    float* __restrict__ tout,                   // [B][C] routing logits
    float* __restrict__ sbuf)                   // [B][C] prior pre-logits
{
    extern __shared__ unsigned short smem[];    // 2 * BUFSZ bf16 = 90 KB

    const int t    = threadIdx.x;
    const int lane = t & 31, w = t >> 5;
    const int wm   = w >> 2, wn = w & 3;
    const int l16  = lane & 15, hh = lane >> 4;
    const int Mb   = blockIdx.x * 64;
    const int Nb   = blockIdx.y * 256;
    const int r    = t >> 2;            // 0..63
    const int cq   = (t & 3) * 16;      // 0,16,32,48
    const int cBase = Mb >> 3;          // first of the block's 8 classes

    // Warm the epilogue operands (global_prefetch_b8).
    __builtin_prefetch(Ainv + (size_t)cBase * 64, 0, 0);
    __builtin_prefetch(xsq + Nb + 64 * wn + l16, 0, 0);

    // Workgroup-relative LDS byte offset of the dynamic-LDS base.
    const unsigned smem_base = (unsigned)(size_t)(void*)smem;

    v8f acc[2][4] = {};

    // Prologue: async-copy chunk 0 into LDS buffer 0.
    issue_chunk(protoB, xnB, Mb, Nb, r, cq, 0,
                smem_base, smem_base + 2u * 64 * LDA);

#pragma unroll 2
    for (int i = 0; i < 8; i++) {
        // My async copies into buf[i&1] have landed; barrier proves everyone's
        // have, and that all waves finished compute(i-1) (last readers of
        // buf[(i+1)&1]) -- so chunk i+1 can be issued immediately and overlap
        // the whole compute of chunk i.
        wait_async0();
        __syncthreads();
        if (i < 7) {
            unsigned nb = smem_base + (unsigned)(((i + 1) & 1) * (2 * BUFSZ));
            issue_chunk(protoB, xnB, Mb, Nb, r, cq, (i + 1) * 64,
                        nb, nb + 2u * 64 * LDA);
        }
        unsigned short* Ab = smem + (i & 1) * BUFSZ;
        unsigned short* Bb = Ab + 64 * LDA;
#pragma unroll
        for (int kk = 0; kk < 64; kk += 32) {
            v16bf af[2], bfr[4];
#pragma unroll
            for (int mt = 0; mt < 2; mt++) {
                int m = 32 * wm + 16 * mt + l16;
                // A 16x32 bf16 layout: lane m; VGPR0-3 K=8h..8h+7, VGPR4-7 K=16+8h..
                af[mt] = mkfrag(&Ab[m * LDA + kk + 8 * hh],
                                &Ab[m * LDA + kk + 16 + 8 * hh]);
            }
#pragma unroll
            for (int nt = 0; nt < 4; nt++) {
                int n = 64 * wn + 16 * nt + l16;
                // B 32x16 bf16 layout: lane n; VGPR0-3 K=8h..8h+7, VGPR4-7 K=16+8h... B[k][n]=xn[n][k].
                bfr[nt] = mkfrag(&Bb[n * LDA + kk + 8 * hh],
                                 &Bb[n * LDA + kk + 16 + 8 * hh]);
            }
#pragma unroll
            for (int mt = 0; mt < 2; mt++)
#pragma unroll
                for (int nt = 0; nt < 4; nt++)
                    acc[mt][nt] = __builtin_amdgcn_wmma_f32_16x16x32_bf16(
                        false, af[mt], false, bfr[nt],
                        (short)0, acc[mt][nt], false, false);
        }
    }

    // ---- Fused epilogue: per-lane dual-ridge solve from the 8-vector h ----
    float tv[2][4], sv[2][4];
#pragma unroll
    for (int mt = 0; mt < 2; mt++) {
        int c = cBase + 4 * wm + 2 * mt + hh;
        float Ai[64];
        const float4* ap = (const float4*)(Ainv + (size_t)c * 64);
#pragma unroll
        for (int i = 0; i < 16; i++) {
            float4 v = ap[i];
            Ai[4 * i + 0] = v.x; Ai[4 * i + 1] = v.y;
            Ai[4 * i + 2] = v.z; Ai[4 * i + 3] = v.w;
        }
#pragma unroll
        for (int nt = 0; nt < 4; nt++) {
            int bb = Nb + 64 * wn + 16 * nt + l16;
            float h8[8];
#pragma unroll
            for (int j = 0; j < 8; j++) h8[j] = acc[mt][nt][j];
            float cross = 0.f, an2 = 0.f, hs = 0.f;
#pragma unroll
            for (int i = 0; i < 8; i++) {
                float a = 0.f;
#pragma unroll
                for (int j = 0; j < 8; j++) a += Ai[i * 8 + j] * h8[j];
                cross += a * h8[i];
                an2   += a * a;
                hs    += h8[i];
            }
            float res = xsq[bb] - cross - RIDGE * an2;
            tv[mt][nt] = res * (-1.f / ROUT_T);
            sv[mt][nt] = hs * (1.f / (8.f * PRIOR_T));
        }
    }

    // ---- LDS transpose so stores become 16B float4 runs (block owns a
    //      dense 256(b) x 8(c) sub-tile of the [B][C] outputs) ----
    float* T = (float*)smem;   // 256*8 floats = 8KB, buffers are dead now
    const int cl = 4 * wm + hh;                 // + 2*mt below
#pragma unroll
    for (int pass = 0; pass < 2; pass++) {      // pass 0: t, pass 1: s
        __syncthreads();
#pragma unroll
        for (int mt = 0; mt < 2; mt++)
#pragma unroll
            for (int nt = 0; nt < 4; nt++) {
                int bl = 64 * wn + 16 * nt + l16;
                T[bl * 8 + cl + 2 * mt] = pass ? sv[mt][nt] : tv[mt][nt];
            }
        __syncthreads();
        float* dst = pass ? sbuf : tout;
#pragma unroll
        for (int half = 0; half < 2; half++) {
            int row = half * 128 + (t >> 1);
            int c4  = (t & 1) * 4;
            float4 v = *(float4*)&T[row * 8 + c4];
            *(float4*)&dst[(size_t)(Nb + row) * CN + cBase + c4] = v;
        }
    }
}

// ---------------------------------------------------------------------------
// Kernel 5: out[b,c] = t[b,c] + log_softmax_c(s[b,:]). One wave per row.
// ---------------------------------------------------------------------------
__global__ __launch_bounds__(256) void k_lsm(const float* __restrict__ sbuf,
                                             float* __restrict__ out) {
    int row  = blockIdx.x * 8 + (threadIdx.x >> 5);
    int lane = threadIdx.x & 31;
    const float* sr = sbuf + (size_t)row * CN;
    float sv[16];
    float mx = -1e30f;
#pragma unroll
    for (int i = 0; i < 16; i++) { sv[i] = sr[lane + 32 * i]; mx = fmaxf(mx, sv[i]); }
#pragma unroll
    for (int m = 16; m >= 1; m >>= 1) mx = fmaxf(mx, __shfl_xor(mx, m, 32));
    float sum = 0.f;
#pragma unroll
    for (int i = 0; i < 16; i++) sum += expf(sv[i] - mx);
#pragma unroll
    for (int m = 16; m >= 1; m >>= 1) sum += __shfl_xor(sum, m, 32);
    float lse = mx + logf(sum);
    float* orow = out + (size_t)row * CN;
#pragma unroll
    for (int i = 0; i < 16; i++) {
        int cidx = lane + 32 * i;
        orow[cidx] = orow[cidx] + sv[i] - lse;
    }
}

// ---------------------------------------------------------------------------
extern "C" void kernel_launch(void* const* d_in, const int* in_sizes, int n_in,
                              void* d_out, int out_size, void* d_ws, size_t ws_size,
                              hipStream_t stream) {
    const float* x     = (const float*)d_in[0];   // [8192,512]
    const float* proto = (const float*)d_in[1];   // [512,8,512]
    float* out = (float*)d_out;                   // [8192,512]

    char* ws = (char*)d_ws;
    unsigned short* protoB = (unsigned short*)(ws);                        // 4 MB
    unsigned short* xnB    = (unsigned short*)(ws + (4u  << 20));          // 8 MB
    float* xsq             = (float*)(ws + (12u << 20));                   // 32 KB
    float* Ainv            = (float*)(ws + (12u << 20) + (64u << 10));     // 128 KB
    float* sbuf            = (float*)(ws + (13u << 20));                   // 16 MB

    // proto fp32 -> bf16 : 2,097,152 elems / 4 per thread = 524,288 threads
    k_cvt <<<2048, 256, 0, stream>>>(proto, protoB);
    // normalize x rows -> bf16 + |xn|^2
    k_norm<<<BN / 8, 256, 0, stream>>>(x, xnB, xsq);
    // per-class 8x8 ridge inverse (fp32)
    k_ainv<<<CN, 64, 0, stream>>>(proto, Ainv);
    // main WMMA GEMM + fused residual/prior epilogue (async double-buffer)
    dim3 g3(4096 / 64, BN / 256);
    k_gemm<<<g3, 256, SMEM_BYTES, stream>>>(protoB, xnB, xsq, Ainv, out, sbuf);
    // row log-softmax of prior, added into logits
    k_lsm <<<BN / 8, 256, 0, stream>>>(sbuf, out);
}
